// GATSBlock_74749610820010
// MI455X (gfx1250) — matrix operand order repaired and checked by reading, not compiled, not run on hardware
//
#include <hip/hip_runtime.h>

typedef __attribute__((ext_vector_type(16))) _Float16 v16h;
typedef __attribute__((ext_vector_type(8)))  _Float16 v8h;
typedef __attribute__((ext_vector_type(8)))  float    v8f;

#define B_   4
#define S_   8192
#define D_   512
#define WS_  512
#define MB_  128      // query rows per block (8 waves x 16)
#define NT_  64       // key-tile size
#define KSTR 520      // Klds row stride (halves): 512 + 8 pad -> conflict-free b128
#define VSTR 72       // VTlds row stride (halves): 64 + 8 pad
#define PSTR 72       // Plds row stride (halves)

#define K_HALVES (NT_ * KSTR)             // 33280 halves = 66560 B per buffer
#define V_HALVES (D_ * VSTR)              // 36864 halves = 73728 B per buffer
#define K_BYTES  (K_HALVES * 2)
#define V_BYTES  (V_HALVES * 2)
#define P_BYTES  (8 * 16 * PSTR * 2)      // 18432
// double-buffered K/V + P : 2*66560 + 2*73728 + 18432 = 299008 B <= 320KB WGP LDS
#define SMEM_BYTES (2 * K_BYTES + 2 * V_BYTES + P_BYTES)

// ds_swizzle group-of-32 XOR patterns: offset = {xor[4:0]<<10 | or<<5 | and[4:0]}
#define SWZ_XOR(x, imm) \
  __int_as_float(__builtin_amdgcn_ds_swizzle(__float_as_int(x), imm))

__device__ __forceinline__ v16h cat8(v8h lo, v8h hi) {
  return __builtin_shufflevector(lo, hi, 0,1,2,3,4,5,6,7,8,9,10,11,12,13,14,15);
}

__global__ __launch_bounds__(256) void gats_local_attn(
    const float* __restrict__ Q, const float* __restrict__ K,
    const float* __restrict__ V, float* __restrict__ O)
{
  extern __shared__ char smem[];
  _Float16* KldsBase  = (_Float16*)(smem);                 // 2 buffers
  _Float16* VTldsBase = (_Float16*)(smem + 2 * K_BYTES);   // 2 buffers
  _Float16* Plds      = (_Float16*)(smem + 2 * K_BYTES + 2 * V_BYTES);

  const int tid   = threadIdx.x;
  const int wave  = tid >> 5;
  const int lane  = tid & 31;
  const int ln    = lane & 15;
  const bool hiH  = lane >= 16;
  const int halfA = hiH ? 8  : 0;   // A-fragment K-dim offset
  const int halfB = hiH ? 8  : 0;   // B-fragment K-dim offset
  const int rowOfs = hiH ? 8 : 0;   // C/D fragment row offset

  const int blk = blockIdx.x;
  const int b   = blk / (S_ / MB_);
  const int m0  = (blk % (S_ / MB_)) * MB_;
  const int wi  = m0 / WS_;
  const int kstart = (wi > 0) ? (wi - 1) * WS_ : 0;
  const int kend   = m0 + MB_;

  const float* Qg = Q + (size_t)b * S_ * D_;
  const float* Kg = K + (size_t)b * S_ * D_;
  const float* Vg = V + (size_t)b * S_ * D_;
  float*       Og = O + (size_t)b * S_ * D_;

  const int qbase = m0 + wave * 16;
  const int qrow  = qbase + ln;
  const int qlast = qbase + 15;

  // ---- staging: K row-major f16, V transposed f16, into buffer `bi`
  auto stage = [&](int j0, int bi) {
    _Float16* Kl = KldsBase  + bi * K_HALVES;
    _Float16* Vl = VTldsBase + bi * V_HALVES;
    #pragma unroll 8
    for (int idx = tid; idx < NT_ * (D_ / 4); idx += 256) {
      const int n = idx >> 7;
      const int d = (idx & 127) << 2;
      float4 kv = *(const float4*)(Kg + (size_t)(j0 + n) * D_ + d);
      _Float16* kd = Kl + n * KSTR + d;
      kd[0] = (_Float16)kv.x; kd[1] = (_Float16)kv.y;
      kd[2] = (_Float16)kv.z; kd[3] = (_Float16)kv.w;
      float4 vv = *(const float4*)(Vg + (size_t)(j0 + n) * D_ + d);
      Vl[(d + 0) * VSTR + n] = (_Float16)vv.x;
      Vl[(d + 1) * VSTR + n] = (_Float16)vv.y;
      Vl[(d + 2) * VSTR + n] = (_Float16)vv.z;
      Vl[(d + 3) * VSTR + n] = (_Float16)vv.w;
    }
  };

  // ---- Preload Q as WMMA A-fragments (f16, pre-scaled by D^-0.5) ----
  const float sc = 0.044194173824159216f;  // 512^-0.5
  v16h qf[16];
  {
    const float* qr = Qg + (size_t)qrow * D_;
    #pragma unroll
    for (int kb = 0; kb < 16; ++kb) {
      const int d0 = kb * 32 + halfA;
      float4 f0 = *(const float4*)(qr + d0);
      float4 f1 = *(const float4*)(qr + d0 + 4);
      float4 f2 = *(const float4*)(qr + d0 + 16);
      float4 f3 = *(const float4*)(qr + d0 + 20);
      v16h t;
      t[0]  = (_Float16)(f0.x * sc); t[1]  = (_Float16)(f0.y * sc);
      t[2]  = (_Float16)(f0.z * sc); t[3]  = (_Float16)(f0.w * sc);
      t[4]  = (_Float16)(f1.x * sc); t[5]  = (_Float16)(f1.y * sc);
      t[6]  = (_Float16)(f1.z * sc); t[7]  = (_Float16)(f1.w * sc);
      t[8]  = (_Float16)(f2.x * sc); t[9]  = (_Float16)(f2.y * sc);
      t[10] = (_Float16)(f2.z * sc); t[11] = (_Float16)(f2.w * sc);
      t[12] = (_Float16)(f3.x * sc); t[13] = (_Float16)(f3.y * sc);
      t[14] = (_Float16)(f3.z * sc); t[15] = (_Float16)(f3.w * sc);
      qf[kb] = t;
    }
  }

  // ---- Output accumulators: 16 rows x 512 cols = 32 C-fragments ----
  v8f o[32];
  #pragma unroll
  for (int dt = 0; dt < 32; ++dt)
    o[dt] = v8f{0.f,0.f,0.f,0.f,0.f,0.f,0.f,0.f};
  // mrow is lane-uniform; lrow is a PER-LANE partial sum (reduced in epilogue)
  float mrow[8], lrow[8];
  #pragma unroll
  for (int r = 0; r < 8; ++r) { mrow[r] = -3.0e38f; lrow[r] = 0.f; }

  _Float16* Pw = Plds + wave * 16 * PSTR;

  // ---- prologue: stage first tile ----
  stage(kstart, 0);
  __syncthreads();

  int buf = 0;
  for (int j0 = kstart; j0 < kend; j0 += NT_) {
    // stage NEXT tile into the other buffer while computing this one
    if (j0 + NT_ < kend) stage(j0 + NT_, buf ^ 1);
    if (j0 + 2 * NT_ < kend) {
      const char* kn = (const char*)(Kg + (size_t)(j0 + 2 * NT_) * D_);
      const char* vn = (const char*)(Vg + (size_t)(j0 + 2 * NT_) * D_);
      #pragma unroll
      for (int t = 0; t < 4; ++t) {
        __builtin_prefetch(kn + (size_t)(tid + t * 256) * 128, 0, 1);
        __builtin_prefetch(vn + (size_t)(tid + t * 256) * 128, 0, 1);
      }
    }

    if (j0 <= qlast) {  // wave-uniform: skip fully-masked tiles
      const _Float16* Klds  = KldsBase  + buf * K_HALVES;
      const _Float16* VTlds = VTldsBase + buf * V_HALVES;

      // ---- S = Q * K^T : kb-outer, 4 independent chains, double-buffered B ----
      v8f s[4];
      #pragma unroll
      for (int nt = 0; nt < 4; ++nt)
        s[nt] = v8f{0.f,0.f,0.f,0.f,0.f,0.f,0.f,0.f};

      v16h bf0[4], bf1[4];
      #pragma unroll
      for (int nt = 0; nt < 4; ++nt) {
        const _Float16* kp = Klds + (nt * 16 + ln) * KSTR + halfB;
        bf0[nt] = cat8(*(const v8h*)kp, *(const v8h*)(kp + 16));
      }
      #pragma unroll
      for (int kb = 0; kb < 16; ++kb) {
        v16h* curb = (kb & 1) ? bf1 : bf0;
        v16h* nxtb = (kb & 1) ? bf0 : bf1;
        if (kb < 15) {
          #pragma unroll
          for (int nt = 0; nt < 4; ++nt) {
            const _Float16* kp = Klds + (nt * 16 + ln) * KSTR + (kb + 1) * 32 + halfB;
            nxtb[nt] = cat8(*(const v8h*)kp, *(const v8h*)(kp + 16));
          }
        }
        #pragma unroll
        for (int nt = 0; nt < 4; ++nt)
          s[nt] = __builtin_amdgcn_wmma_f32_16x16x32_f16(
                      false, qf[kb], false, curb[nt], (short)0, s[nt], false, false);
      }

      // ---- row max (causal mask only on the single diagonal tile) ----
      float rmax[8];
      if (j0 + NT_ - 1 > qbase) {   // wave-uniform: tile overlaps query rows
        #pragma unroll
        for (int r = 0; r < 8; ++r) {
          const int rowa = qbase + r + rowOfs;
          float mx = -3.0e38f;
          #pragma unroll
          for (int nt = 0; nt < 4; ++nt) {
            const int col = j0 + nt * 16 + ln;
            float v = s[nt][r];
            v = (col <= rowa) ? v : -3.0e38f;
            s[nt][r] = v;
            mx = fmaxf(mx, v);
          }
          rmax[r] = mx;
        }
      } else {                      // fully-valid tile: plain max
        #pragma unroll
        for (int r = 0; r < 8; ++r) {
          rmax[r] = fmaxf(fmaxf(s[0][r], s[1][r]), fmaxf(s[2][r], s[3][r]));
        }
      }
      #pragma unroll
      for (int r = 0; r < 8; ++r) {  // 16-lane XOR max-reduce via ds_swizzle
        rmax[r] = fmaxf(rmax[r], SWZ_XOR(rmax[r], 0x041F));
        rmax[r] = fmaxf(rmax[r], SWZ_XOR(rmax[r], 0x081F));
        rmax[r] = fmaxf(rmax[r], SWZ_XOR(rmax[r], 0x101F));
        rmax[r] = fmaxf(rmax[r], SWZ_XOR(rmax[r], 0x201F));
      }

      // ---- online softmax update (l kept per-lane, no reduction here) ----
      float scl[8];
      #pragma unroll
      for (int r = 0; r < 8; ++r) {
        const float mn = fmaxf(mrow[r], rmax[r]);
        scl[r]  = __expf(mrow[r] - mn);
        mrow[r] = mn;
        float sum = 0.f;
        #pragma unroll
        for (int nt = 0; nt < 4; ++nt) {
          const float p = __expf(s[nt][r] - mn);
          s[nt][r] = p;
          sum += p;
        }
        lrow[r] = lrow[r] * scl[r] + sum;  // scl is lane-uniform -> linear
      }
      v8f sclv;
      #pragma unroll
      for (int r = 0; r < 8; ++r) sclv[r] = scl[r];

      // ---- P -> LDS (D-layout -> row-major) ----
      #pragma unroll
      for (int r = 0; r < 8; ++r) {
        const int rowM = r + rowOfs;
        #pragma unroll
        for (int nt = 0; nt < 4; ++nt)
          Pw[rowM * PSTR + nt * 16 + ln] = (_Float16)s[nt][r];
      }

      // ---- load P as A-fragments (2 x 32-key blocks) ----
      v16h pa[2];
      #pragma unroll
      for (int jb = 0; jb < 2; ++jb) {
        const _Float16* pp = Pw + ln * PSTR + jb * 32 + halfA;
        pa[jb] = cat8(*(const v8h*)pp, *(const v8h*)(pp + 16));
      }

      // ---- O = O*scale + P*V : double-buffered V fragments, fused rescale ----
      v16h bv0[2], bv1[2];
      #pragma unroll
      for (int jb = 0; jb < 2; ++jb) {
        const _Float16* vp = VTlds + (0 * 16 + ln) * VSTR + jb * 32 + halfB;
        bv0[jb] = cat8(*(const v8h*)vp, *(const v8h*)(vp + 16));
      }
      #pragma unroll
      for (int dt = 0; dt < 32; ++dt) {
        v16h* curv = (dt & 1) ? bv1 : bv0;
        v16h* nxtv = (dt & 1) ? bv0 : bv1;
        if (dt < 31) {
          #pragma unroll
          for (int jb = 0; jb < 2; ++jb) {
            const _Float16* vp = VTlds + ((dt + 1) * 16 + ln) * VSTR + jb * 32 + halfB;
            nxtv[jb] = cat8(*(const v8h*)vp, *(const v8h*)(vp + 16));
          }
        }
        o[dt] = o[dt] * sclv;  // v_pk_mul_f32, co-executes with WMMA
        o[dt] = __builtin_amdgcn_wmma_f32_16x16x32_f16(
                    false, pa[0], false, curv[0], (short)0, o[dt], false, false);
        o[dt] = __builtin_amdgcn_wmma_f32_16x16x32_f16(
                    false, pa[1], false, curv[1], (short)0, o[dt], false, false);
      }
    }
    __syncthreads();  // compute done on buf, staging done on buf^1
    buf ^= 1;
  }

  // ---- epilogue: reduce l across the 16-lane halves, normalize, store ----
  v8f invv;
  #pragma unroll
  for (int r = 0; r < 8; ++r) {
    float l = lrow[r];
    l += SWZ_XOR(l, 0x041F);
    l += SWZ_XOR(l, 0x081F);
    l += SWZ_XOR(l, 0x101F);
    l += SWZ_XOR(l, 0x201F);
    invv[r] = 1.0f / l;
  }
  #pragma unroll
  for (int dt = 0; dt < 32; ++dt) {
    v8f res = o[dt] * invv;
    #pragma unroll
    for (int r = 0; r < 8; ++r) {
      const int rowa = qbase + r + rowOfs;
      Og[(size_t)rowa * D_ + dt * 16 + ln] = res[r];
    }
  }
}

extern "C" void kernel_launch(void* const* d_in, const int* in_sizes, int n_in,
                              void* d_out, int out_size, void* d_ws, size_t ws_size,
                              hipStream_t stream) {
  (void)in_sizes; (void)n_in; (void)out_size; (void)d_ws; (void)ws_size;
  const float* q = (const float*)d_in[0];  // text  -> Q
  const float* k = (const float*)d_in[1];  // audio -> K
  const float* v = (const float*)d_in[2];  // video -> V
  float* out = (float*)d_out;
  dim3 grid(B_ * S_ / MB_);   // 256 blocks
  dim3 block(256);            // 8 waves (wave32), 1 block per WGP (299KB LDS)
  gats_local_attn<<<grid, block, SMEM_BYTES, stream>>>(q, k, v, out);
}
